// StructureExtractor_51161650430437
// MI455X (gfx1250) — matrix operand order repaired and checked by reading, not compiled, not run on hardware
//
#include <hip/hip_runtime.h>
#include <hip/hip_bf16.h>

// ---------------------------------------------------------------------------
// Problem constants (from reference): N=8, L=16384, C=256, K=64, d_in=512
// MLP: 512 -> 128 (relu) -> 256 (relu) -> 256
// ---------------------------------------------------------------------------
#define NB    8
#define LPTS  16384
#define CDIM  256
#define KANC  64
#define DIN   512
#define HD1   128
#define HD2   256
#define TM    128          // rows per block tile

typedef __attribute__((ext_vector_type(16))) __bf16 bf16x16;
typedef __attribute__((ext_vector_type(8)))  float  f32x8;

// Native bf16 converts (lower to v_cvt_*bf16* on gfx1250)
__device__ __forceinline__ unsigned short f2bf(float f) {
    __bf16 h = (__bf16)f;
    return __builtin_bit_cast(unsigned short, h);
}
__device__ __forceinline__ float bf2f(unsigned short u) {
    return (float)__builtin_bit_cast(__bf16, u);
}

union ABfrag { bf16x16 v; uint4 q[2]; };

// ---------------------------------------------------------------------------
// Weight prep: f32 [K][N] row-major  ->  bf16 transposed [N][K] in workspace
// ---------------------------------------------------------------------------
__global__ void prep_weights(const float* __restrict__ W1,
                             const float* __restrict__ W2,
                             const float* __restrict__ W3,
                             unsigned short* __restrict__ wt1,
                             unsigned short* __restrict__ wt2,
                             unsigned short* __restrict__ wt3) {
    int i = blockIdx.x * blockDim.x + threadIdx.x;    // 65536 threads
    if (i < HD1 * DIN)  wt1[i] = f2bf(W1[(i & (DIN - 1)) * HD1 + (i >> 9)]);
    if (i < HD2 * HD1)  wt2[i] = f2bf(W2[(i & (HD1 - 1)) * HD2 + (i >> 7)]);
    if (i < HD2 * HD2)  wt3[i] = f2bf(W3[(i & (HD2 - 1)) * HD2 + (i >> 8)]);
}

// ---------------------------------------------------------------------------
// Fused: structured feature + concat + 3-layer MLP, bf16 WMMA, f32 accumulate
// grid = 2 (sets) * 8 (batch) * 128 (row tiles) = 2048 blocks, 256 threads
// Loop order: K outermost with persistent accumulators so each B fragment is
// fetched once per wave and reused across all 8 M-tiles (L2 weight traffic
// = 320 KB per block).
// ---------------------------------------------------------------------------
__global__ __launch_bounds__(256, 1) void fused_structure_mlp(
    const float* __restrict__ feat0, const float* __restrict__ feat1,
    const float* __restrict__ pts0,  const float* __restrict__ pts1,
    const int*   __restrict__ anci,  const int*   __restrict__ ancj,
    const float* __restrict__ b1,    const float* __restrict__ b2,
    const float* __restrict__ b3,
    const unsigned short* __restrict__ wt1,
    const unsigned short* __restrict__ wt2,
    const unsigned short* __restrict__ wt3,
    float* __restrict__ out) {

    __shared__ __align__(16) unsigned short sX [TM * DIN];   // 128 KB
    __shared__ __align__(16) unsigned short sH1[TM * HD1];   //  32 KB
    __shared__ __align__(16) unsigned short sH2[TM * HD2];   //  64 KB

    const int bid  = blockIdx.x;
    const int set  = bid >> 10;              // 0: (feat0,pts0,anchor_i) 1: other
    const int tile = bid & 1023;
    const int n    = tile >> 7;              // batch index
    const int row0 = (tile & 127) * TM;      // first point of this tile
    const int tid  = threadIdx.x;

    const float* feat = set ? feat1 : feat0;
    const float* pts  = set ? pts1  : pts0;
    const int*   anc  = set ? ancj  : anci;

    // ---- stage feat tile (f32 -> bf16), fully coalesced float4 loads -------
    const float4* fbase = (const float4*)(feat + ((size_t)n * LPTS + row0) * CDIM);
    for (int e = tid; e < TM * CDIM / 4; e += 256) {
        int row = e >> 6;                    // 64 float4 per row
        int cg  = (e & 63) << 2;
        float4 f = fbase[e];
        uint2 pk;
        pk.x = (unsigned)f2bf(f.x) | ((unsigned)f2bf(f.y) << 16);
        pk.y = (unsigned)f2bf(f.z) | ((unsigned)f2bf(f.w) << 16);
        *(uint2*)(sX + row * DIN + cg) = pk;
    }

    // ---- structured feature: 1 thread per row ------------------------------
    if (tid < TM) {
        const int row = tid;
        const size_t gp = ((size_t)n * LPTS + row0 + row) * 3;
        const float px = pts[gp], py = pts[gp + 1], pz = pts[gp + 2];
        float s0 = 0.f, s1 = 0.f, s2 = 0.f, s3 = 0.f;
        unsigned short* srow = sX + row * DIN + CDIM;
        for (int k = 0; k < KANC; ++k) {
            const int idx = anc[n * KANC + k];
            const size_t ap = ((size_t)n * LPTS + idx) * 3;
            const float dx = px - pts[ap];
            const float dy = py - pts[ap + 1];
            const float dz = pz - pts[ap + 2];
            const float dd = dx * dx + dy * dy + dz * dz;
            srow[0 * KANC + k] = f2bf(dx); s0 += fabsf(dx);
            srow[1 * KANC + k] = f2bf(dy); s1 += fabsf(dy);
            srow[2 * KANC + k] = f2bf(dz); s2 += fabsf(dz);
            srow[3 * KANC + k] = f2bf(dd); s3 += dd;
        }
        const float i0 = 1.f / s0, i1 = 1.f / s1, i2 = 1.f / s2, i3 = 1.f / s3;
        for (int k = 0; k < KANC; ++k) {
            srow[0 * KANC + k] = f2bf(bf2f(srow[0 * KANC + k]) * i0);
            srow[1 * KANC + k] = f2bf(bf2f(srow[1 * KANC + k]) * i1);
            srow[2 * KANC + k] = f2bf(bf2f(srow[2 * KANC + k]) * i2);
            srow[3 * KANC + k] = f2bf(bf2f(srow[3 * KANC + k]) * i3);
        }
    }
    __syncthreads();

    const int lane = tid & 31;
    const int wave = tid >> 5;
    const int c16  = lane & 15;          // column within 16-wide N tile
    const int hf   = lane >> 4;          // half-wave select

    // ---- GEMM1: x[128,512] @ W1 -> h1[128,128]  (1 N-tile per wave) --------
    {
        const int col = wave * 16 + c16;
        const float bias = b1[col];
        const uint4* bbase = (const uint4*)(wt1 + (size_t)col * DIN);
        f32x8 acc[8] = {};
        #pragma unroll 1
        for (int ks = 0; ks < DIN / 32; ++ks) {
            ABfrag b;
            const uint4* bp = bbase + ((ks * 32 + hf * 8) >> 3);
            b.q[0] = bp[0];
            b.q[1] = bp[2];
            #pragma unroll
            for (int m = 0; m < TM / 16; ++m) {
                ABfrag a;
                const uint4* ap =
                    (const uint4*)(sX + (m * 16 + c16) * DIN + ks * 32 + hf * 8);
                a.q[0] = ap[0];          // K = base .. base+7
                a.q[1] = ap[2];          // K = base+16 .. base+23
                acc[m] = __builtin_amdgcn_wmma_f32_16x16x32_bf16(
                    false, a.v, false, b.v, (short)0, acc[m], false, false);
            }
        }
        #pragma unroll
        for (int m = 0; m < TM / 16; ++m) {
            #pragma unroll
            for (int v = 0; v < 8; ++v) {
                const int row = m * 16 + hf * 8 + v;
                sH1[row * HD1 + col] = f2bf(fmaxf(acc[m][v] + bias, 0.f));
            }
        }
    }
    __syncthreads();

    // ---- GEMM2: h1[128,128] @ W2 -> h2[128,256]  (2 N-tiles per wave) ------
    for (int t = 0; t < 2; ++t) {
        const int col = (wave * 2 + t) * 16 + c16;
        const float bias = b2[col];
        const uint4* bbase = (const uint4*)(wt2 + (size_t)col * HD1);
        f32x8 acc[8] = {};
        #pragma unroll 1
        for (int ks = 0; ks < HD1 / 32; ++ks) {
            ABfrag b;
            const uint4* bp = bbase + ((ks * 32 + hf * 8) >> 3);
            b.q[0] = bp[0];
            b.q[1] = bp[2];
            #pragma unroll
            for (int m = 0; m < TM / 16; ++m) {
                ABfrag a;
                const uint4* ap =
                    (const uint4*)(sH1 + (m * 16 + c16) * HD1 + ks * 32 + hf * 8);
                a.q[0] = ap[0];
                a.q[1] = ap[2];
                acc[m] = __builtin_amdgcn_wmma_f32_16x16x32_bf16(
                    false, a.v, false, b.v, (short)0, acc[m], false, false);
            }
        }
        #pragma unroll
        for (int m = 0; m < TM / 16; ++m) {
            #pragma unroll
            for (int v = 0; v < 8; ++v) {
                const int row = m * 16 + hf * 8 + v;
                sH2[row * HD2 + col] = f2bf(fmaxf(acc[m][v] + bias, 0.f));
            }
        }
    }
    __syncthreads();

    // ---- GEMM3: h2[128,256] @ W3 -> out[128,256]  (2 N-tiles per wave) -----
    float* outp = out + (size_t)set * ((size_t)NB * LPTS * CDIM)
                      + ((size_t)n * LPTS + row0) * CDIM;
    for (int t = 0; t < 2; ++t) {
        const int col = (wave * 2 + t) * 16 + c16;
        const float bias = b3[col];
        const uint4* bbase = (const uint4*)(wt3 + (size_t)col * HD2);
        f32x8 acc[8] = {};
        #pragma unroll 1
        for (int ks = 0; ks < HD2 / 32; ++ks) {
            ABfrag b;
            const uint4* bp = bbase + ((ks * 32 + hf * 8) >> 3);
            b.q[0] = bp[0];
            b.q[1] = bp[2];
            #pragma unroll
            for (int m = 0; m < TM / 16; ++m) {
                ABfrag a;
                const uint4* ap =
                    (const uint4*)(sH2 + (m * 16 + c16) * HD2 + ks * 32 + hf * 8);
                a.q[0] = ap[0];
                a.q[1] = ap[2];
                acc[m] = __builtin_amdgcn_wmma_f32_16x16x32_bf16(
                    false, a.v, false, b.v, (short)0, acc[m], false, false);
            }
        }
        #pragma unroll
        for (int m = 0; m < TM / 16; ++m) {
            #pragma unroll
            for (int v = 0; v < 8; ++v) {
                const int row = m * 16 + hf * 8 + v;
                outp[(size_t)row * CDIM + col] = acc[m][v] + bias;
            }
        }
    }
}

// ---------------------------------------------------------------------------
extern "C" void kernel_launch(void* const* d_in, const int* in_sizes, int n_in,
                              void* d_out, int out_size, void* d_ws, size_t ws_size,
                              hipStream_t stream) {
    const float* feat0 = (const float*)d_in[0];
    const float* feat1 = (const float*)d_in[1];
    const float* pts0  = (const float*)d_in[2];
    const float* pts1  = (const float*)d_in[3];
    const int*   anci  = (const int*)d_in[4];
    const int*   ancj  = (const int*)d_in[5];
    const float* W1    = (const float*)d_in[6];
    const float* b1    = (const float*)d_in[7];
    const float* W2    = (const float*)d_in[8];
    const float* b2    = (const float*)d_in[9];
    const float* W3    = (const float*)d_in[10];
    const float* b3    = (const float*)d_in[11];
    float* out = (float*)d_out;

    // bf16 transposed weights in workspace: 320 KB total
    unsigned short* wt1 = (unsigned short*)d_ws;           // [128][512]
    unsigned short* wt2 = wt1 + (size_t)HD1 * DIN;         // [256][128]
    unsigned short* wt3 = wt2 + (size_t)HD2 * HD1;         // [256][256]

    prep_weights<<<256, 256, 0, stream>>>(W1, W2, W3, wt1, wt2, wt3);

    const int blocks = 2 * NB * (LPTS / TM);               // 2048
    fused_structure_mlp<<<blocks, 256, 0, stream>>>(
        feat0, feat1, pts0, pts1, anci, ancj, b1, b2, b3, wt1, wt2, wt3, out);
}
